// AttentionUnit_16441134809156
// MI455X (gfx1250) — matrix operand order repaired and checked by reading, not compiled, not run on hardware
//
#include <hip/hip_runtime.h>
#include <hip/hip_bf16.h>

typedef __bf16        v16bf __attribute__((ext_vector_type(16)));
typedef float         v8f   __attribute__((ext_vector_type(8)));
typedef unsigned int  u32x4 __attribute__((ext_vector_type(4)));
typedef int           i32x4 __attribute__((ext_vector_type(4)));
typedef int           i32x8 __attribute__((ext_vector_type(8)));

union ABFrag { v16bf bf; u32x4 q[2]; };

__device__ __forceinline__ unsigned f2bf(float x) {
    union { float f; unsigned u; } v; v.f = x;
    unsigned r = v.u + 0x7FFFu + ((v.u >> 16) & 1u);
    return r >> 16;   // bf16 bits in low 16
}

// LDS byte offset of a generic pointer known to point into LDS
__device__ __forceinline__ unsigned lds_addr_of(void* p) {
    return (unsigned)(unsigned long long)(__attribute__((address_space(3))) void*)p;
}

#define EDIM 1024
#define SDIM 2048

// ---------------------------------------------------------------------------
// Projection: Y[m][n] = sum_e X[m][e] * W[n][e] + bias[n], output bf16.
// transpose==1 stores Y as [B][E][S] (for V) so later B-fragments are linear.
// Block: 256 threads, tile M=64 x N=64, BK=32. One wmma per tile per K-step.
// ---------------------------------------------------------------------------
__global__ __launch_bounds__(256) void proj_kernel(
    const float* __restrict__ X, const float* __restrict__ W,
    const float* __restrict__ bias, unsigned short* __restrict__ Y,
    int transpose)
{
    __shared__ unsigned short Xs[64 * 32];
    __shared__ unsigned short Ws[64 * 32];

    const int t    = threadIdx.x;
    const int lane = t & 31;
    const int wave = t >> 5;
    const int m0   = blockIdx.x * 64;
    const int n0   = blockIdx.y * 64;

    const int mi  = wave & 3;          // 0..3 : 16-row sub-tile
    const int nio = (wave >> 2) * 2;   // 0 or 2 : two 16-col sub-tiles

    v8f acc0 = {}; v8f acc1 = {};

    const int lrow = t >> 2;           // 0..63
    const int lcol = (t & 3) * 8;      // 0,8,16,24 (bf16 elems)

    for (int k0 = 0; k0 < EDIM; k0 += 32) {
        // cooperative fp32 load -> bf16 -> LDS
        {
            const float4* xr = (const float4*)(X + (size_t)(m0 + lrow) * EDIM + k0 + lcol);
            const float4* wr = (const float4*)(W + (size_t)(n0 + lrow) * EDIM + k0 + lcol);
            float4 xa = xr[0], xb = xr[1];
            float4 wa = wr[0], wb = wr[1];
            u32x4 xv, wv;
            xv[0] = f2bf(xa.x) | (f2bf(xa.y) << 16);
            xv[1] = f2bf(xa.z) | (f2bf(xa.w) << 16);
            xv[2] = f2bf(xb.x) | (f2bf(xb.y) << 16);
            xv[3] = f2bf(xb.z) | (f2bf(xb.w) << 16);
            wv[0] = f2bf(wa.x) | (f2bf(wa.y) << 16);
            wv[1] = f2bf(wa.z) | (f2bf(wa.w) << 16);
            wv[2] = f2bf(wb.x) | (f2bf(wb.y) << 16);
            wv[3] = f2bf(wb.z) | (f2bf(wb.w) << 16);
            *(u32x4*)&Xs[lrow * 32 + lcol] = xv;
            *(u32x4*)&Ws[lrow * 32 + lcol] = wv;
        }
        __syncthreads();

        // A fragment (16x32 bf16): lane<16 -> row=lane, K{0..7,16..23}
        ABFrag a;
        {
            const int r  = mi * 16 + (lane & 15);
            const int ko = (lane < 16) ? 0 : 8;
            a.q[0] = *(const u32x4*)&Xs[r * 32 + ko];
            a.q[1] = *(const u32x4*)&Xs[r * 32 + ko + 16];
        }
        // B fragments (32x16 bf16): lane<16 -> K{0..7,16..23}, else K{8..15,24..31}
        ABFrag b0, b1;
        {
            const int ko = (lane < 16) ? 0 : 8;
            const int n1 = (nio + 0) * 16 + (lane & 15);
            const int n2 = (nio + 1) * 16 + (lane & 15);
            b0.q[0] = *(const u32x4*)&Ws[n1 * 32 + ko];
            b0.q[1] = *(const u32x4*)&Ws[n1 * 32 + ko + 16];
            b1.q[0] = *(const u32x4*)&Ws[n2 * 32 + ko];
            b1.q[1] = *(const u32x4*)&Ws[n2 * 32 + ko + 16];
        }
        acc0 = __builtin_amdgcn_wmma_f32_16x16x32_bf16(false, a.bf, false, b0.bf,
                                                       (short)0, acc0, false, false);
        acc1 = __builtin_amdgcn_wmma_f32_16x16x32_bf16(false, a.bf, false, b1.bf,
                                                       (short)0, acc1, false, false);
        __syncthreads();
    }

    // bias + bf16 store (C layout: VGPR r -> row r (+8 for upper lanes))
    const int gn0  = n0 + (nio + 0) * 16 + (lane & 15);
    const int gn1  = n0 + (nio + 1) * 16 + (lane & 15);
    const float bb0 = bias[gn0], bb1 = bias[gn1];
    const int mrow = m0 + mi * 16 + ((lane < 16) ? 0 : 8);
#pragma unroll
    for (int r = 0; r < 8; ++r) {
        const int gm = mrow + r;
        const unsigned short o0 = (unsigned short)f2bf(acc0[r] + bb0);
        const unsigned short o1 = (unsigned short)f2bf(acc1[r] + bb1);
        if (!transpose) {
            Y[(size_t)gm * EDIM + gn0] = o0;
            Y[(size_t)gm * EDIM + gn1] = o1;
        } else {
            const int bidx = gm >> 11, s = gm & 2047;   // S = 2048
            Y[((size_t)bidx * EDIM + gn0) * SDIM + s] = o0;
            Y[((size_t)bidx * EDIM + gn1) * SDIM + s] = o1;
        }
    }
}

// ---------------------------------------------------------------------------
// Causal attention: one block per (batch, 16-row q tile).
// LDS: q tile bf16 [16][1024] (32KB, TDM-loaded) + scores f32 [16][2048]
// (128KB) + P bf16 [16][2048] (64KB) = 224KB of the 320KB WGP LDS.
// ---------------------------------------------------------------------------
__global__ __launch_bounds__(256) void attn_kernel(
    const unsigned short* __restrict__ qp,   // [B*S][E] bf16
    const unsigned short* __restrict__ kp,   // [B*S][E] bf16
    const unsigned short* __restrict__ vpT,  // [B][E][S] bf16
    float* __restrict__ out)                 // [B*S][E] f32
{
    extern __shared__ char smem[];
    unsigned short* qs = (unsigned short*)smem;                                 // 16*1024 bf16
    float* sc = (float*)(smem + (size_t)16 * EDIM * sizeof(unsigned short));    // 16*2048 f32
    unsigned short* ps = (unsigned short*)(smem + (size_t)16 * EDIM * 2
                                                + (size_t)16 * SDIM * 4);       // 16*2048 bf16

    __shared__ float red[16][17];
    __shared__ float rowstat[16];

    const int t    = threadIdx.x;
    const int lane = t & 31;
    const int wave = t >> 5;
    const int qt   = blockIdx.x;            // 0..127
    const int b    = blockIdx.y;            // 0..3
    const int qbase = b * SDIM + qt * 16;

    // ---- stage q tile via Tensor Data Mover (one DMA, issued by wave 0) ----
    if (wave == 0) {
        const unsigned long long ga =
            (unsigned long long)(const void*)(qp + (size_t)qbase * EDIM);
        const unsigned ldsa = lds_addr_of(qs);
        u32x4 g0;
        g0[0] = 1u;                                   // count=1 user descriptor
        g0[1] = ldsa;                                 // lds_addr (bytes)
        g0[2] = (unsigned)(ga & 0xFFFFFFFFu);         // global_addr[31:0]
        g0[3] = (unsigned)((ga >> 32) & 0x01FFFFFFu)  // global_addr[56:32]
              | (2u << 30);                           // type=2 ("image")
        i32x8 g1;
        g1[0] = (int)(1u << 16);                      // wg_mask=0, data_size=1 (2B)
        g1[1] = (int)((unsigned)EDIM << 16);          // tensor_dim0[15:0]
        g1[2] = (int)(8192u << 16);                   // dim0[31:16]=0 | tensor_dim1[15:0]=8192
        g1[3] = (int)((unsigned)EDIM << 16);          // dim1[31:16]=0 | tile_dim0=1024
        g1[4] = 16;                                   // tile_dim1=16, tile_dim2=0
        g1[5] = EDIM;                                 // tensor_dim0_stride[31:0]
        g1[6] = 0;                                    // stride hi | dim1_stride lo
        g1[7] = 0;
        i32x4 gz4 = {0, 0, 0, 0};
        i32x8 gz8 = {0, 0, 0, 0, 0, 0, 0, 0};
        __builtin_amdgcn_tensor_load_to_lds(g0, g1, gz4, gz4, gz8, 0);
        __builtin_amdgcn_s_wait_tensorcnt(0);
    }
    __syncthreads();

    const int kmax = (qt / 8 + 1) * 128;    // causal: key blocks beyond are all masked

    // ---------------- Phase 1: S = Q K^T / sqrt(E), masked, into LDS ----------
    {
        const int arow = lane & 15;
        const int ako  = (lane < 16) ? 0 : 8;
        const int bko  = (lane < 16) ? 0 : 8;
        for (int kb = 0; kb < kmax; kb += 128) {
            const int key = kb + wave * 16 + (lane & 15);
            const unsigned short* krow = kp + (size_t)(b * SDIM + key) * EDIM;
            v8f s = {};
            for (int e0 = 0; e0 < EDIM; e0 += 32) {
                __builtin_prefetch((const void*)(krow + e0 + 256), 0, 1);
                ABFrag a, bb;
                a.q[0]  = *(const u32x4*)&qs[arow * EDIM + e0 + ako];
                a.q[1]  = *(const u32x4*)&qs[arow * EDIM + e0 + ako + 16];
                bb.q[0] = *(const u32x4*)&krow[e0 + bko];
                bb.q[1] = *(const u32x4*)&krow[e0 + bko + 16];
                s = __builtin_amdgcn_wmma_f32_16x16x32_bf16(false, a.bf, false, bb.bf,
                                                            (short)0, s, false, false);
            }
            const int mbase = (lane < 16) ? 0 : 8;
#pragma unroll
            for (int r = 0; r < 8; ++r) {
                const int m = mbase + r;
                float v = s[r] * 0.03125f;            // 1/sqrt(1024)
                if (key > qt * 16 + m) v = -1e30f;    // causal mask
                sc[m * SDIM + key] = v;
            }
        }
    }
    __syncthreads();

    // -------- Phase 2: row softmax over [0, kmax), emit P as bf16 ------------
    {
        const int row = t >> 4;     // 0..15
        const int l16 = t & 15;
        float mx = -1e30f;
        for (int k = l16; k < kmax; k += 16) mx = fmaxf(mx, sc[row * SDIM + k]);
        red[row][l16] = mx;
        __syncthreads();
        if (l16 == 0) {
            float m2 = red[row][0];
            for (int i = 1; i < 16; ++i) m2 = fmaxf(m2, red[row][i]);
            rowstat[row] = m2;
        }
        __syncthreads();
        const float rm = rowstat[row];
        float sum = 0.f;
        for (int k = l16; k < kmax; k += 16) {
            const float e = __expf(sc[row * SDIM + k] - rm);
            sc[row * SDIM + k] = e;
            sum += e;
        }
        red[row][l16] = sum;
        __syncthreads();
        if (l16 == 0) {
            float s2 = 0.f;
            for (int i = 0; i < 16; ++i) s2 += red[row][i];
            rowstat[row] = 1.0f / s2;
        }
        __syncthreads();
        const float inv = rowstat[row];
        for (int k = l16; k < kmax; k += 16)
            ps[row * SDIM + k] = (unsigned short)f2bf(sc[row * SDIM + k] * inv);
    }
    __syncthreads();

    // ---------------- Phase 3: O = P V ; wave owns a 128-wide E strip ---------
    {
        const int nb    = wave * 128;
        const int arow  = lane & 15;
        const int ako   = (lane < 16) ? 0 : 8;
        const int bko   = (lane < 16) ? 0 : 8;
        v8f acc[8] = {};
        for (int k0 = 0; k0 < kmax; k0 += 32) {
            ABFrag a;
            a.q[0] = *(const u32x4*)&ps[arow * SDIM + k0 + ako];
            a.q[1] = *(const u32x4*)&ps[arow * SDIM + k0 + ako + 16];
#pragma unroll
            for (int nt = 0; nt < 8; ++nt) {
                const int gn = nb + nt * 16 + (lane & 15);
                const unsigned short* vr =
                    vpT + ((size_t)b * EDIM + gn) * SDIM + k0 + bko;
                ABFrag bb;
                bb.q[0] = *(const u32x4*)vr;
                bb.q[1] = *(const u32x4*)(vr + 16);
                acc[nt] = __builtin_amdgcn_wmma_f32_16x16x32_bf16(false, a.bf, false, bb.bf,
                                                                  (short)0, acc[nt], false, false);
            }
        }
        const int mbase = (lane < 16) ? 0 : 8;
#pragma unroll
        for (int nt = 0; nt < 8; ++nt) {
            const int gn = nb + nt * 16 + (lane & 15);
#pragma unroll
            for (int r = 0; r < 8; ++r) {
                out[(size_t)(qbase + mbase + r) * EDIM + gn] = acc[nt][r];
            }
        }
    }
}

// ---------------------------------------------------------------------------
extern "C" void kernel_launch(void* const* d_in, const int* in_sizes, int n_in,
                              void* d_out, int out_size, void* d_ws, size_t ws_size,
                              hipStream_t stream) {
    (void)in_sizes; (void)n_in; (void)out_size; (void)ws_size;
    const float* q  = (const float*)d_in[0];
    const float* v  = (const float*)d_in[1];
    const float* k  = (const float*)d_in[2];
    const float* Wq = (const float*)d_in[3];
    const float* bq = (const float*)d_in[4];
    const float* Wv = (const float*)d_in[5];
    const float* bv = (const float*)d_in[6];
    const float* Wk = (const float*)d_in[7];
    const float* bk = (const float*)d_in[8];

    unsigned short* qpw  = (unsigned short*)d_ws;                 // [8192][1024] bf16
    unsigned short* kpw  = qpw + (size_t)8192 * 1024;             // [8192][1024] bf16
    unsigned short* vpTw = kpw + (size_t)8192 * 1024;             // [4][1024][2048] bf16

    dim3 pgrid(128, 16), pblock(256);
    proj_kernel<<<pgrid, pblock, 0, stream>>>(q, Wq, bq, qpw, 0);
    proj_kernel<<<pgrid, pblock, 0, stream>>>(k, Wk, bk, kpw, 0);
    proj_kernel<<<pgrid, pblock, 0, stream>>>(v, Wv, bv, vpTw, 1);

    const size_t smem = (size_t)16 * EDIM * sizeof(unsigned short)   // q tile
                      + (size_t)16 * SDIM * sizeof(float)            // scores
                      + (size_t)16 * SDIM * sizeof(unsigned short);  // P bf16
    dim3 agrid(128, 4);
    attn_kernel<<<agrid, dim3(256), smem, stream>>>(qpw, kpw, vpTw, (float*)d_out);
}
